// SingleHexagonLoss_75161927680564
// MI455X (gfx1250) — matrix operand order repaired and checked by reading, not compiled, not run on hardware
//
#include <hip/hip_runtime.h>
#include <hip/hip_bf16.h>

// ---------------------------------------------------------------------------
// SingleHexagonLoss for MI455X (gfx1250, wave32, WMMA)
//
// preds:  [16,1,128,128] f32,  hex_template: [1,1,128,128] f32 -> scalar f32
//
// mean(conv_same(binary, T)) == (1/16384) * sum_p binary[p]*W[p] with
// W[p] = SAT-window sum of T over [p-64, p+63]^2 (XLA SAME pad_low=63).
// Weighted sum + per-image area run as a 16x16384 x 16384x3 GEMM on
// v_wmma_f32_16x16x32_f16 (f16 exact: binary is 0/1, W split hi/lo).
// W is staged to LDS with global_load_async_to_lds_b128 when available.
// ---------------------------------------------------------------------------

typedef __attribute__((ext_vector_type(16))) _Float16 v16h;
typedef __attribute__((ext_vector_type(8)))  float    v8f;
typedef __attribute__((ext_vector_type(16))) float    v16f;
typedef int v4i_b128 __attribute__((vector_size(16)));   // matches builtin param

#define NB   16          // batch
#define HW   16384       // 128*128
#define SATP 129         // LDS row stride (bank-conflict padding)

#if defined(__gfx1250__) && __has_builtin(__builtin_amdgcn_global_load_async_to_lds_b128) && __has_builtin(__builtin_amdgcn_s_wait_asynccnt)
#define HEX_ASYNC_LDS 1
#else
#define HEX_ASYNC_LDS 0
#endif

// ---------------------------------------------------------------------------
// Kernel A: summed-area table of template (in LDS) -> per-pixel weight W,
//           and zero the accumulator block.
// acc layout (floats): [0:16) tsum, [16:32) area, [32:48) perim, [48:64) multi
// ---------------------------------------------------------------------------
__global__ __launch_bounds__(128) void hex_compute_w_kernel(
    const float* __restrict__ tmpl, float* __restrict__ W,
    float* __restrict__ acc) {
  __shared__ float sat[128 * SATP];
  const int t = threadIdx.x;  // 0..127

  for (int x = 0; x < 128; ++x) sat[t * SATP + x] = tmpl[t * 128 + x];
  __syncthreads();
  float run = 0.0f;  // row prefix (thread t scans row t)
  for (int x = 0; x < 128; ++x) { run += sat[t * SATP + x]; sat[t * SATP + x] = run; }
  __syncthreads();
  run = 0.0f;        // column prefix (thread t scans column t)
  for (int y = 0; y < 128; ++y) { run += sat[y * SATP + t]; sat[y * SATP + t] = run; }
  __syncthreads();

  const int py = t;
  int y0 = py - 64; if (y0 < 0) y0 = 0;
  int y1 = py + 63; if (y1 > 127) y1 = 127;
  for (int px = 0; px < 128; ++px) {
    int x0 = px - 64; if (x0 < 0) x0 = 0;
    int x1 = px + 63; if (x1 > 127) x1 = 127;
    float s = sat[y1 * SATP + x1];
    if (x0 > 0)           s -= sat[y1 * SATP + (x0 - 1)];
    if (y0 > 0)           s -= sat[(y0 - 1) * SATP + x1];
    if (x0 > 0 && y0 > 0) s += sat[(y0 - 1) * SATP + (x0 - 1)];
    W[py * 128 + px] = s;
  }
  if (t < 64) acc[t] = 0.0f;   // zero accumulators every call
}

// ---------------------------------------------------------------------------
// Kernel B: WMMA weighted reduction. 8 blocks x 8 waves; wave covers 256 px.
//   A[M=image][K=pixel] = binary (0/1 f16, exact)
//   B[K][0]=W_hi, B[K][1]=1, B[K][2]=W_lo (hi/lo f16 split of integer W)
//   D[i][0]+D[i][2] = sum(binary*W), D[i][1] = area
// B is built branch-free with per-lane bit masks (no EXEC divergence in the
// K-loop -> wmma legal and no cmpx/branch overhead).
// ---------------------------------------------------------------------------
__global__ __launch_bounds__(256) void hex_wmma_reduce_kernel(
    const float* __restrict__ preds, const float* __restrict__ W,
    float* __restrict__ tsum, float* __restrict__ area) {
  const int lane = threadIdx.x & 31;
  const int wave = threadIdx.x >> 5;
  const int kb0  = blockIdx.x * 2048;       // block's K slice
  const int kblk = kb0 + wave * 256;        // wave's K slice
  const int M    = lane & 15;
  const bool hiL = lane >= 16;
  const float* __restrict__ img = preds + M * HW;

  // Per-lane column select masks for B (branch-free).
  const unsigned int selhi  = (M == 0) ? 0xFFFFu : 0u;
  const unsigned int selone = (M == 1) ? 0x3C00u : 0u;   // f16 1.0 pre-masked
  const unsigned int sello  = (M == 2) ? 0xFFFFu : 0u;
  const int abase = hiL ? 16 : 0;   // A: lanes 0-15 K=0..15; hi K=16..31
  const int wbase = hiL ? 16 : 0;   // B: lanes 0-15 rows 0..15; hi rows 16..31

  // Stage this block's W slice (2048 floats = 8KB) into LDS.
  __shared__ __align__(16) float lds_w[2048];
#if HEX_ASYNC_LDS
  {
    // 512 b128 transfers / 256 threads = 2 per thread, ASYNCcnt-tracked.
    for (int i = threadIdx.x; i < 512; i += 256) {
      const float* g = W + kb0 + i * 4;
      float* l = &lds_w[i * 4];
      __builtin_amdgcn_global_load_async_to_lds_b128(
          (__attribute__((address_space(1))) v4i_b128*)(uintptr_t)g,
          (__attribute__((address_space(3))) v4i_b128*)(unsigned int)(uintptr_t)l,
          0, 0);
    }
    __builtin_amdgcn_s_wait_asynccnt(0);
    __syncthreads();
  }
#else
  {
    for (int i = threadIdx.x; i < 2048; i += 256) lds_w[i] = W[kb0 + i];
    __syncthreads();
  }
#endif

  v8f c = {};
#pragma unroll
  for (int it = 0; it < 8; ++it) {
    const int k0  = kblk + it * 32;
    const int lk0 = (wave * 256) + it * 32;   // offset inside lds_w
    __builtin_prefetch(img + k0 + 256, 0, 1);

    // A operand: two contiguous 8-float runs -> wide loads, then binarize.
    const v8f r0 = *(const v8f*)(img + k0 + abase);        // K: abase+0..7
    const v8f r1 = *(const v8f*)(img + k0 + abase + 8);    // K: abase+8..15
    v16h a;
#pragma unroll
    for (int j = 0; j < 8; ++j) {
      a[j]     = (r0[j] > 0.5f) ? (_Float16)1.0f : (_Float16)0.0f;
      a[j + 8] = (r1[j] > 0.5f) ? (_Float16)1.0f : (_Float16)0.0f;
    }

    // B operand: 16 contiguous W values from LDS, branch-free column select.
    const v16f wv = *(const v16f*)(&lds_w[lk0 + wbase]);
    v16h b;
#pragma unroll
    for (int j = 0; j < 16; ++j) {
      const float    w   = wv[j];
      const _Float16 whi = (_Float16)w;
      const _Float16 wlo = (_Float16)(w - (float)whi);   // in {-1,0,1}: exact
      const unsigned int bits =
          (((unsigned int)__builtin_bit_cast(unsigned short, whi)) & selhi) |
          selone |
          (((unsigned int)__builtin_bit_cast(unsigned short, wlo)) & sello);
      b[j] = __builtin_bit_cast(_Float16, (unsigned short)bits);
    }

    // (neg_a, A, neg_b, B, c_mod, C, reuse_a, reuse_b)
    c = __builtin_amdgcn_wmma_f32_16x16x32_f16(false, a, false, b, (short)0, c,
                                               false, false);
  }

  const int mo = hiL ? 8 : 0;   // C/D: VGPR r -> M=r (lanes 0-15) / M=r+8
  if (M == 0 || M == 2) {
#pragma unroll
    for (int r = 0; r < 8; ++r) atomicAdd(&tsum[r + mo], c[r]);
  } else if (M == 1) {
#pragma unroll
    for (int r = 0; r < 8; ++r) atomicAdd(&area[r + mo], c[r]);
  }
}

// ---------------------------------------------------------------------------
// Kernel C: Sobel perimeter. edges>0.5 <=> gx!=0 || gy!=0 (binary inputs).
// ---------------------------------------------------------------------------
__global__ __launch_bounds__(256) void hex_perimeter_kernel(
    const float* __restrict__ preds, float* __restrict__ perim) {
  const int b = blockIdx.x;
  const float* __restrict__ img = preds + b * HW;
  __shared__ int s_cnt;
  if (threadIdx.x == 0) s_cnt = 0;
  __syncthreads();

  int cnt = 0;
  for (int p = threadIdx.x; p < HW; p += 256) {
    const int py = p >> 7, px = p & 127;
    int v[3][3];
#pragma unroll
    for (int dy = -1; dy <= 1; ++dy)
#pragma unroll
      for (int dx = -1; dx <= 1; ++dx) {
        const int y = py + dy, x = px + dx;
        v[dy + 1][dx + 1] =
            (y >= 0 && y < 128 && x >= 0 && x < 128) ? (img[y * 128 + x] > 0.5f ? 1 : 0) : 0;
      }
    const int gx = -v[0][0] + v[0][2] - 2 * v[1][0] + 2 * v[1][2] - v[2][0] + v[2][2];
    const int gy = -v[0][0] - 2 * v[0][1] - v[0][2] + v[2][0] + 2 * v[2][1] + v[2][2];
    if (gx != 0 || gy != 0) ++cnt;
  }
  atomicAdd(&s_cnt, cnt);
  __syncthreads();
  if (threadIdx.x == 0) perim[b] = (float)s_cnt;
}

// ---------------------------------------------------------------------------
// Kernel D: 4-connected component count, LDS-resident min-label propagation.
// Gauss-Seidel sweeps reach the same fixed point as the reference Jacobi loop.
// ---------------------------------------------------------------------------
__global__ __launch_bounds__(256) void hex_components_kernel(
    const float* __restrict__ preds, float* __restrict__ multi) {
  const int b = blockIdx.x;
  const float* __restrict__ img = preds + b * HW;
  __shared__ unsigned short lab[HW];
  __shared__ int s_changed, s_count;
  const int t = threadIdx.x;
  const unsigned short BIGL = 0xFFFFu;

  for (int p = t; p < HW; p += 256)
    lab[p] = (img[p] > 0.5f) ? (unsigned short)p : BIGL;
  if (t == 0) s_count = 0;
  __syncthreads();

  const int p0 = t * 64, p1 = p0 + 64;
  for (;;) {
    if (t == 0) s_changed = 0;
    __syncthreads();
    bool ch = false;
    for (int p = p0; p < p1; ++p) {      // forward sweep
      unsigned short l = lab[p];
      if (l == BIGL) continue;
      unsigned short m = l;
      const int py = p >> 7, px = p & 127;
      if (py > 0)   { unsigned short q = lab[p - 128]; if (q < m) m = q; }
      if (py < 127) { unsigned short q = lab[p + 128]; if (q < m) m = q; }
      if (px > 0)   { unsigned short q = lab[p - 1];   if (q < m) m = q; }
      if (px < 127) { unsigned short q = lab[p + 1];   if (q < m) m = q; }
      if (m < l) { lab[p] = m; ch = true; }
    }
    for (int p = p1 - 1; p >= p0; --p) { // backward sweep
      unsigned short l = lab[p];
      if (l == BIGL) continue;
      unsigned short m = l;
      const int py = p >> 7, px = p & 127;
      if (py > 0)   { unsigned short q = lab[p - 128]; if (q < m) m = q; }
      if (py < 127) { unsigned short q = lab[p + 128]; if (q < m) m = q; }
      if (px > 0)   { unsigned short q = lab[p - 1];   if (q < m) m = q; }
      if (px < 127) { unsigned short q = lab[p + 1];   if (q < m) m = q; }
      if (m < l) { lab[p] = m; ch = true; }
    }
    if (ch) s_changed = 1;
    __syncthreads();
    const int cont = s_changed;   // read between barriers: race-free, uniform
    __syncthreads();
    if (!cont) break;
  }

  int cnt = 0;
  for (int p = t; p < HW; p += 256)
    if (lab[p] == (unsigned short)p) ++cnt;   // background is BIGL (never == p)
  atomicAdd(&s_count, cnt);
  __syncthreads();
  if (t == 0) {
    int mc = s_count - 1;
    if (mc < 0) mc = 0;
    multi[b] = (float)mc;
  }
}

// ---------------------------------------------------------------------------
// Kernel E: combine per-image stats into the scalar loss.
// ---------------------------------------------------------------------------
__global__ __launch_bounds__(32) void hex_finalize_kernel(
    const float* __restrict__ acc, float* __restrict__ out) {
  if (threadIdx.x == 0) {
    float loss = 0.0f;
    for (int b = 0; b < NB; ++b) {
      const float ts = acc[b], ar = acc[16 + b], pe = acc[32 + b], mc = acc[48 + b];
      const float geom = (ar < 1e-6f) ? 0.0f : fabsf(pe * pe / (ar + 1e-6f) - 4.51f);
      const float tl   = 1.0f - ts * (1.0f / 16384.0f);
      loss += mc + geom + tl;
    }
    out[0] = loss * (1.0f / 16.0f);
  }
}

// ---------------------------------------------------------------------------
extern "C" void kernel_launch(void* const* d_in, const int* in_sizes, int n_in,
                              void* d_out, int out_size, void* d_ws, size_t ws_size,
                              hipStream_t stream) {
  (void)in_sizes; (void)n_in; (void)out_size; (void)ws_size;
  const float* preds = (const float*)d_in[0];   // [16,1,128,128] f32
  const float* tmpl  = (const float*)d_in[1];   // [1,1,128,128]  f32
  float* out = (float*)d_out;

  float* W   = (float*)d_ws;        // 16384 floats
  float* acc = W + HW;              // 64 floats: tsum|area|perim|multi

  hex_compute_w_kernel <<<1, 128, 0, stream>>>(tmpl, W, acc);
  hex_perimeter_kernel <<<NB, 256, 0, stream>>>(preds, acc + 32);
  hex_components_kernel<<<NB, 256, 0, stream>>>(preds, acc + 48);
  hex_wmma_reduce_kernel<<<8, 256, 0, stream>>>(preds, W, acc + 0, acc + 16);
  hex_finalize_kernel  <<<1, 32, 0, stream>>>(acc, out);
}
